// DecoderBlockMoE_54520314855564
// MI455X (gfx1250) — matrix operand order repaired and checked by reading, not compiled, not run on hardware
//
#include <hip/hip_runtime.h>

// ---------------- static config (matches reference) ----------------
#define BB   2
#define SS   2048
#define DD   1024
#define HH   16
#define HDIM 64
#define ROT  32
#define CONT 32
#define LQ   512
#define LKV  256
#define FF   1024
#define NR   7
#define CAP  585
#define CPAD 640          // CAP padded to multiple of 128 (GEMM tile M)
#define TT   (BB*SS)      // 4096 tokens

typedef __attribute__((ext_vector_type(16))) _Float16 v16h;
typedef __attribute__((ext_vector_type(8)))  float    v8f;

// ---------------- WMMA GEMM: C[M,N](f32) = A[M,K](f16) * B(f16) ----------------
// A: [M,K] row-major halves.
// BT=false: B is K-pair-packed dwords: Bp[(k/2)*N + n] = {B[k][n], B[k+1][n]}.
// BT=true : B is [N,K] row-major halves (pairs along K are contiguous dwords).
// Block: 256 threads = 8 waves (4 in M x 2 in N); block tile 128(M) x 64(N).
// Each wave computes 32x32 (4 accumulators, 4 WMMAs/K-step). Double-buffered LDS,
// 128-bit global staging. REQUIRES: M%128==0, N%64==0, K%32==0 (all shapes here comply).
template <bool BT>
__global__ void __launch_bounds__(256)
gemm_wmma(const _Float16* __restrict__ A, const unsigned int* __restrict__ B32,
          float* __restrict__ C, int M, int N, int K, int ldc) {
  __shared__ __align__(16) unsigned int la[2][128 * 16];  // [m][pk] (pk = K-pair 0..15)
  __shared__ __align__(16) unsigned int lb[2][16 * 64];   // [pk][n]
  const int tid  = threadIdx.x;
  const int m0   = blockIdx.y * 128;
  const int n0   = blockIdx.x * 64;
  const int lane = tid & 31;
  const int wave = tid >> 5;
  const int wm   = wave >> 1;          // 0..3  (32-row band)
  const int wn   = wave & 1;           // 0..1  (32-col band)
  const int hi   = lane >> 4;          // K-band select (0/1)
  const int r    = lane & 15;
  const int K2   = K >> 1;
  const unsigned int* A32 = (const unsigned int*)A;

  auto stage = [&](int kt, int bufi) {
    const int k0h = kt * 16;           // K-pair offset
    // ---- issue ALL global loads first (distinct regs -> clauseable) ----
    const int pa0 = 4 * tid;           // A: 128 rows x 16 pairs; 2x uint4 per thread
    const int pa1 = 4 * tid + 1024;
    const uint4 a0 = *(const uint4*)(A32 + (size_t)(m0 + (pa0 >> 4)) * K2 + k0h + (pa0 & 15));
    const uint4 a1 = *(const uint4*)(A32 + (size_t)(m0 + (pa1 >> 4)) * K2 + k0h + (pa1 & 15));
    uint4 bv;
    if (BT) {
      const int tn = tid & 63, tk = tid >> 6;   // 4 consecutive pk's, fixed col
      bv = *(const uint4*)(B32 + (size_t)(n0 + tn) * K2 + k0h + tk * 4);
    } else {
      const int p = 4 * tid;           // B: 16 pairs x 64 cols; uint4 per thread
      bv = *(const uint4*)(B32 + (size_t)(k0h + (p >> 6)) * N + (p & 63));
    }
    // ---- then all LDS stores ----
    *(uint4*)(&la[bufi][pa0]) = a0;
    *(uint4*)(&la[bufi][pa1]) = a1;
    if (BT) {
      const int tn = tid & 63, tk = tid >> 6;
      lb[bufi][(tk * 4 + 0) * 64 + tn] = bv.x;
      lb[bufi][(tk * 4 + 1) * 64 + tn] = bv.y;
      lb[bufi][(tk * 4 + 2) * 64 + tn] = bv.z;
      lb[bufi][(tk * 4 + 3) * 64 + tn] = bv.w;
    } else {
      *(uint4*)(&lb[bufi][4 * tid]) = bv;
    }
  };

  v8f acc00 = {0.f, 0.f, 0.f, 0.f, 0.f, 0.f, 0.f, 0.f};
  v8f acc01 = acc00, acc10 = acc00, acc11 = acc00;

  const int nk = K >> 5;               // K / 32 steps
  stage(0, 0);
  __syncthreads();
  int buf = 0;
  for (int kt = 0; kt < nk; ++kt) {
    if (kt + 1 < nk) stage(kt + 1, buf ^ 1);   // prefetch next tile into alt buffer

    union { v16h v; unsigned int u[8]; } fa0, fa1, fb0, fb1;
    const int arow0 = (wm * 32 + r) * 16;
    const int arow1 = arow0 + 256;             // +16 rows
    const int pbase = hi * 4;
#pragma unroll
    for (int i = 0; i < 4; ++i) {
      // A fragment: lanes 0-15 -> K {0..7,16..23}; lanes 16-31 -> K {8..15,24..31}
      fa0.u[i]     = la[buf][arow0 + pbase + i];
      fa0.u[4 + i] = la[buf][arow0 + pbase + 8 + i];
      fa1.u[i]     = la[buf][arow1 + pbase + i];
      fa1.u[4 + i] = la[buf][arow1 + pbase + 8 + i];
    }
    const int bcol = wn * 32 + r;
#pragma unroll
    for (int j = 0; j < 8; ++j) {
      // B fragment: lanes 0-15 -> K {0..7,16..23}; lanes 16-31 -> K {8..15,24..31}
      fb0.u[j] = lb[buf][(hi * 4 + j + (j & 4)) * 64 + bcol];
      fb1.u[j] = lb[buf][(hi * 4 + j + (j & 4)) * 64 + bcol + 16];
    }
    acc00 = __builtin_amdgcn_wmma_f32_16x16x32_f16(false, fa0.v, false, fb0.v,
                                                   (short)0, acc00, false, false);
    acc01 = __builtin_amdgcn_wmma_f32_16x16x32_f16(false, fa0.v, false, fb1.v,
                                                   (short)0, acc01, false, false);
    acc10 = __builtin_amdgcn_wmma_f32_16x16x32_f16(false, fa1.v, false, fb0.v,
                                                   (short)0, acc10, false, false);
    acc11 = __builtin_amdgcn_wmma_f32_16x16x32_f16(false, fa1.v, false, fb1.v,
                                                   (short)0, acc11, false, false);
    __syncthreads();
    buf ^= 1;
  }
#pragma unroll
  for (int v = 0; v < 8; ++v) {
    const int mA = m0 + wm * 32 + v + hi * 8;  // first 16-row band of wave tile
    const int mB = mA + 16;                    // second 16-row band
    const int nA = n0 + wn * 32 + r;
    C[(size_t)mA * ldc + nA]      = acc00[v];
    C[(size_t)mA * ldc + nA + 16] = acc01[v];
    C[(size_t)mB * ldc + nA]      = acc10[v];
    C[(size_t)mB * ldc + nA + 16] = acc11[v];
  }
}

// ---------------- elementwise / helper kernels ----------------
__global__ void __launch_bounds__(256)
cvt_f2h(const float* __restrict__ a, _Float16* __restrict__ o, int n) {
  int i = blockIdx.x * 256 + threadIdx.x;
  if (i < n) o[i] = (_Float16)a[i];
}

// pack f32 [K,N] -> K-pair dwords: o[(k/2)*N+n] = {h(B[k][n]), h(B[k+1][n])}
__global__ void __launch_bounds__(256)
packB_f2h(const float* __restrict__ B, unsigned int* __restrict__ o, int K, int N) {
  long long i = (long long)blockIdx.x * 256 + threadIdx.x;
  long long tot = (long long)(K >> 1) * N;
  if (i >= tot) return;
  int n = (int)(i % N);
  long long pk = i / N;
  union { _Float16 h[2]; unsigned int u; } x;
  x.h[0] = (_Float16)B[(size_t)(2 * pk) * N + n];
  x.h[1] = (_Float16)B[(size_t)(2 * pk + 1) * N + n];
  o[i] = x.u;
}

__global__ void __launch_bounds__(256)
zero_f(float* __restrict__ o, int n) {
  int i = blockIdx.x * 256 + threadIdx.x;
  if (i < n) o[i] = 0.f;
}

__global__ void __launch_bounds__(256)
add2_f(const float* __restrict__ a, const float* __restrict__ b,
       float* __restrict__ o, int n) {
  int i = blockIdx.x * 256 + threadIdx.x;
  if (i < n) o[i] = a[i] + b[i];
}

__global__ void __launch_bounds__(256)
add3_f(const float* __restrict__ a, const float* __restrict__ b,
       const float* __restrict__ c, float* __restrict__ o, int n) {
  int i = blockIdx.x * 256 + threadIdx.x;
  if (i < n) o[i] = a[i] + b[i] + c[i];
}

// RMSNorm row -> f16 (one block per row)
__global__ void __launch_bounds__(256)
rmsnorm_h(const float* __restrict__ x, const float* __restrict__ w,
          _Float16* __restrict__ o) {
  __shared__ float red[256];
  const int row = blockIdx.x;
  const float* xr = x + (size_t)row * DD;
  float s = 0.f;
  for (int d = threadIdx.x; d < DD; d += 256) { float v = xr[d]; s += v * v; }
  red[threadIdx.x] = s; __syncthreads();
  for (int off = 128; off > 0; off >>= 1) {
    if (threadIdx.x < off) red[threadIdx.x] += red[threadIdx.x + off];
    __syncthreads();
  }
  const float rinv = rsqrtf(red[0] / (float)DD + 1e-6f);
  for (int d = threadIdx.x; d < DD; d += 256)
    o[(size_t)row * DD + d] = (_Float16)(xr[d] * rinv * w[d]);
}

// Assemble q or k in [B,H,S,HD] f16: first 32 dims = cont, last 32 = RoPE(rot)
__global__ void __launch_bounds__(256)
build_qk(const float* __restrict__ cont, const float* __restrict__ rot,
         _Float16* __restrict__ out) {
  int i = blockIdx.x * 256 + threadIdx.x;     // over B*H*S*HD = 4.19M
  if (i >= BB * HH * SS * HDIM) return;
  int d = i & 63, s = (i >> 6) & (SS - 1), h = (i >> 17) & 15, b = i >> 21;
  int t = b * SS + s;
  float val;
  if (d < CONT) {
    val = cont[(size_t)t * DD + h * HDIM + d];
  } else {
    int j  = d - CONT;                        // 0..31
    int jj = j & 15;
    float invf = expf(-(float)jj * 0.0625f * 9.210340371976184f); // 10000^(-jj/16)
    float ang = (float)s * invf;
    float c = cosf(ang), sn = sinf(ang);
    const float* rr = rot + (size_t)t * (HH * 2 * ROT) + h * (2 * ROT);
    float xr = rr[j];
    float rh = (j < 16) ? -rr[j + 16] : rr[j - 16];
    val = xr * c + rh * sn;
  }
  out[i] = (_Float16)val;
}

// v: [T, D] f32 -> per-(b,h) K-pair-packed dwords: vp[bh][s/2][d] = {v[s], v[s+1]}
__global__ void __launch_bounds__(256)
build_v(const float* __restrict__ vc, unsigned int* __restrict__ vp) {
  int i = blockIdx.x * 256 + threadIdx.x;     // over BB*HH*(SS/2)*HDIM = 2.10M
  if (i >= BB * HH * (SS / 2) * HDIM) return;
  int n  = i & 63;
  int pk = (i >> 6) & (SS / 2 - 1);
  int bh = i >> 16;
  int h = bh & 15, b = bh >> 4;
  int t0 = b * SS + 2 * pk;
  union { _Float16 hh[2]; unsigned int u; } x;
  x.hh[0] = (_Float16)vc[(size_t)t0 * DD + h * HDIM + n];
  x.hh[1] = (_Float16)vc[(size_t)(t0 + 1) * DD + h * HDIM + n];
  vp[i] = x.u;
}

// causal softmax over one S x S score tile (scale 1/sqrt(HD), mask -1e9) -> f16
__global__ void __launch_bounds__(256)
softmax_causal(const float* __restrict__ sc, _Float16* __restrict__ at) {
  __shared__ float red[256];
  const int q = blockIdx.x;
  const size_t base = (size_t)q * SS;
  float loc[SS / 256];
  float mx = -3.4e38f;
#pragma unroll
  for (int i = 0; i < SS / 256; ++i) {
    int k = threadIdx.x + i * 256;
    float v = sc[base + k] * 0.125f + (k > q ? -1e9f : 0.f);
    loc[i] = v; mx = fmaxf(mx, v);
  }
  red[threadIdx.x] = mx; __syncthreads();
  for (int off = 128; off > 0; off >>= 1) {
    if (threadIdx.x < off) red[threadIdx.x] = fmaxf(red[threadIdx.x], red[threadIdx.x + off]);
    __syncthreads();
  }
  mx = red[0]; __syncthreads();
  float sum = 0.f;
#pragma unroll
  for (int i = 0; i < SS / 256; ++i) { loc[i] = expf(loc[i] - mx); sum += loc[i]; }
  red[threadIdx.x] = sum; __syncthreads();
  for (int off = 128; off > 0; off >>= 1) {
    if (threadIdx.x < off) red[threadIdx.x] += red[threadIdx.x + off];
    __syncthreads();
  }
  const float inv = 1.f / red[0];
#pragma unroll
  for (int i = 0; i < SS / 256; ++i)
    at[base + threadIdx.x + i * 256] = (_Float16)(loc[i] * inv);
}

// SwiGLU: h[M,2*FF] f32 -> o[M,FF] f16;  o = h1 * silu(h2)
__global__ void __launch_bounds__(256)
swiglu_h(const float* __restrict__ h, _Float16* __restrict__ o, int M) {
  int i = blockIdx.x * 256 + threadIdx.x;
  if (i >= M * FF) return;
  int m = i / FF, f = i - m * FF;
  float h1 = h[(size_t)m * (2 * FF) + f];
  float h2 = h[(size_t)m * (2 * FF) + FF + f];
  o[i] = (_Float16)(h1 * h2 / (1.f + expf(-h2)));
}

// gate: aff = sigmoid(xn2 @ Wgate + bias), plus per-token top-2 expert ids
__global__ void __launch_bounds__(256)
gate_topk(const _Float16* __restrict__ xn2, const float* __restrict__ Wg,
          const float* __restrict__ bias, float* __restrict__ aff,
          int* __restrict__ top2) {
  __shared__ float red[256];
  __shared__ float affs[8];
  const int t = blockIdx.x;
  float acc[NR] = {0.f, 0.f, 0.f, 0.f, 0.f, 0.f, 0.f};
  for (int d = threadIdx.x; d < DD; d += 256) {
    float xv = (float)xn2[(size_t)t * DD + d];
#pragma unroll
    for (int e = 0; e < NR; ++e) acc[e] += xv * Wg[d * NR + e];
  }
  for (int e = 0; e < NR; ++e) {
    red[threadIdx.x] = acc[e]; __syncthreads();
    for (int off = 128; off > 0; off >>= 1) {
      if (threadIdx.x < off) red[threadIdx.x] += red[threadIdx.x + off];
      __syncthreads();
    }
    if (threadIdx.x == 0) affs[e] = 1.f / (1.f + expf(-(red[0] + bias[e])));
    __syncthreads();
  }
  if (threadIdx.x == 0) {
    int i1 = 0; float v1 = affs[0];
    for (int e = 1; e < NR; ++e) if (affs[e] > v1) { v1 = affs[e]; i1 = e; }
    int i2 = 0; float v2 = -1e30f;
    for (int e = 0; e < NR; ++e) if (e != i1 && affs[e] > v2) { v2 = affs[e]; i2 = e; }
    for (int e = 0; e < NR; ++e) aff[t * NR + e] = affs[e];
    top2[t * 2] = i1; top2[t * 2 + 1] = i2;
  }
}

// per-expert capacity selection: bitonic sort 4096 (member ? aff : -inf) desc, keep CAP
__global__ void __launch_bounds__(1024)
select_capacity(const float* __restrict__ aff, const int* __restrict__ top2,
                float* __restrict__ wsel, int* __restrict__ sidx) {
  __shared__ float key[TT];
  __shared__ int   vid[TT];
  const int e = blockIdx.x;
  for (int i = threadIdx.x; i < TT; i += 1024) {
    bool mem = (top2[i * 2] == e) || (top2[i * 2 + 1] == e);
    key[i] = mem ? aff[i * NR + e] : -1e30f;
    vid[i] = i;
  }
  __syncthreads();
  for (int k = 2; k <= TT; k <<= 1) {
    for (int j = k >> 1; j > 0; j >>= 1) {
      for (int i = threadIdx.x; i < TT; i += 1024) {
        int ixj = i ^ j;
        if (ixj > i) {
          bool up = ((i & k) == 0);   // descending overall
          bool sw = up ? (key[i] < key[ixj]) : (key[i] > key[ixj]);
          if (sw) {
            float tk = key[i]; key[i] = key[ixj]; key[ixj] = tk;
            int   tv = vid[i]; vid[i] = vid[ixj]; vid[ixj] = tv;
          }
        }
      }
      __syncthreads();
    }
  }
  for (int c = threadIdx.x; c < CAP; c += 1024) {
    float v = key[c];
    wsel[e * CAP + c] = (v > -1e29f) ? v : 0.f;
    sidx[e * CAP + c] = vid[c];
  }
}

// gather tokens into padded [NR, CPAD, D] f16 expert batches
__global__ void __launch_bounds__(256)
gather_h(const _Float16* __restrict__ xn2, const int* __restrict__ sidx,
         _Float16* __restrict__ g) {
  long long i = (long long)blockIdx.x * 256 + threadIdx.x;
  if (i >= (long long)NR * CPAD * DD) return;
  int d = (int)(i % DD);
  long long rem = i / DD;
  int c = (int)(rem % CPAD);
  int e = (int)(rem / CPAD);
  _Float16 v = (_Float16)0.f;
  if (c < CAP) v = xn2[(size_t)sidx[e * CAP + c] * DD + d];
  g[i] = v;
}

// per-expert scatter: routed[token] += eout[c] * w[c]  (tokens unique within expert)
__global__ void __launch_bounds__(256)
scatter_e(const float* __restrict__ eout, const int* __restrict__ sidx,
          const float* __restrict__ wsel, float* __restrict__ routed, int e) {
  int i = blockIdx.x * 256 + threadIdx.x;
  if (i >= CAP * DD) return;
  int c = i / DD, d = i - c * DD;
  float w = wsel[e * CAP + c];
  int t = sidx[e * CAP + c];
  routed[(size_t)t * DD + d] += eout[(size_t)c * DD + d] * w;
}

// ---------------- host orchestration ----------------
extern "C" void kernel_launch(void* const* d_in, const int* in_sizes, int n_in,
                              void* d_out, int out_size, void* d_ws, size_t ws_size,
                              hipStream_t stream) {
  (void)in_sizes; (void)n_in; (void)out_size; (void)ws_size;
  const float* x        = (const float*)d_in[0];
  /* d_in[1] causal_mask: implied by kernel */
  const float* Wq_lat   = (const float*)d_in[2];
  const float* Wkv_lat  = (const float*)d_in[3];
  const float* Wrot_q   = (const float*)d_in[4];
  const float* Wrot_k   = (const float*)d_in[5];
  const float* Wq_up    = (const float*)d_in[6];
  const float* Wk_up    = (const float*)d_in[7];
  const float* Wv_up    = (const float*)d_in[8];
  const float* Wout     = (const float*)d_in[9];
  const float* norm1_w  = (const float*)d_in[10];
  const float* norm2_w  = (const float*)d_in[11];
  const float* Ws1      = (const float*)d_in[12];
  const float* Ws2      = (const float*)d_in[13];
  const float* Wr1      = (const float*)d_in[14];
  const float* Wr2      = (const float*)d_in[15];
  const float* Wgate    = (const float*)d_in[16];
  const float* ebias    = (const float*)d_in[17];
  float* out = (float*)d_out;

  char* p = (char*)d_ws;
  auto alloc = [&](size_t bytes) -> void* {
    void* r = (void*)p; p += (bytes + 255) & ~(size_t)255; return r;
  };
  auto H = [&](size_t n) { return (_Float16*)alloc(n * sizeof(_Float16)); };
  auto U = [&](size_t n) { return (unsigned int*)alloc(n * sizeof(unsigned int)); };
  auto F = [&](size_t n) { return (float*)alloc(n * sizeof(float)); };

  // K-pair-packed f16 weights ((K/2)*N dwords each)
  unsigned int* wq_lat_p = U((size_t)(DD / 2) * LQ);
  unsigned int* wkv_lat_p= U((size_t)(DD / 2) * LKV);
  unsigned int* wrot_q_p = U((size_t)(LQ / 2) * (HH * 2 * ROT));
  unsigned int* wrot_k_p = U((size_t)(LKV / 2) * (HH * 2 * ROT));
  unsigned int* wq_up_p  = U((size_t)(LQ / 2) * DD);
  unsigned int* wk_up_p  = U((size_t)(LKV / 2) * DD);
  unsigned int* wv_up_p  = U((size_t)(LKV / 2) * DD);
  unsigned int* wout_p   = U((size_t)(DD / 2) * DD);
  unsigned int* ws1_p    = U((size_t)(DD / 2) * (2 * FF));
  unsigned int* ws2_p    = U((size_t)(FF / 2) * DD);
  unsigned int* wr1_p    = U((size_t)NR * (DD / 2) * (2 * FF));
  unsigned int* wr2_p    = U((size_t)NR * (FF / 2) * DD);
  // activations
  _Float16* xn_h   = H((size_t)TT * DD);
  float*    zq_f   = F((size_t)TT * LQ);
  _Float16* zq_h   = H((size_t)TT * LQ);
  float*    zkv_f  = F((size_t)TT * LKV);
  _Float16* zkv_h  = H((size_t)TT * LKV);
  float*    qr_f   = F((size_t)TT * HH * 2 * ROT);   // 16MB, reused as scores later
  float*    kr_f   = F((size_t)TT * HH * 2 * ROT);   // 16MB, reused as aout later
  float*    qc_f   = F((size_t)TT * DD);             // reused as proj later
  float*    kc_f   = F((size_t)TT * DD);             // reused as xmid later
  float*    vc_f   = F((size_t)TT * DD);
  _Float16* q_h    = H((size_t)BB * HH * SS * HDIM);
  _Float16* k_h    = H((size_t)BB * HH * SS * HDIM);
  unsigned int* v_p = U((size_t)BB * HH * (SS / 2) * HDIM);  // K-pair-packed V
  _Float16* attn_h = H((size_t)SS * SS);
  _Float16* ao_h   = H((size_t)TT * DD);
  _Float16* xn2_h  = H((size_t)TT * DD);
  float*    hbig_f = F((size_t)TT * 2 * FF);
  _Float16* hsw_h  = H((size_t)TT * FF);
  float*    shared_f = F((size_t)TT * DD);
  float*    routed_f = F((size_t)TT * DD);
  float*    eout_f   = F((size_t)CPAD * DD);
  float*    aff_f    = F((size_t)TT * NR);
  int*      top2_i   = (int*)alloc((size_t)TT * 2 * sizeof(int));
  float*    wsel_f   = F((size_t)NR * CAP);
  int*      sidx_i   = (int*)alloc((size_t)NR * CAP * sizeof(int));
  _Float16* g_h      = H((size_t)NR * CPAD * DD);    // padded expert batches
  // aliases over dead buffers (free after q/k/v are built)
  float* scores_f = qr_f;   // S*S
  float* aout_f   = kr_f;   // T*D
  float* proj_f   = qc_f;   // T*D
  float* xmid_f   = kc_f;   // T*D

  auto cvt = [&](const float* s, _Float16* d, size_t n) {
    cvt_f2h<<<dim3((unsigned)((n + 255) / 256)), dim3(256), 0, stream>>>(s, d, (int)n);
  };
  auto packB = [&](const float* s, unsigned int* d, int K, int N) {
    long long n = (long long)(K / 2) * N;
    packB_f2h<<<dim3((unsigned)((n + 255) / 256)), dim3(256), 0, stream>>>(s, d, K, N);
  };
  auto gemm = [&](const _Float16* A, const unsigned int* Bm, float* Cm,
                  int M, int N, int K, int ldc, bool bt) {
    dim3 g((unsigned)(N / 64), (unsigned)(M / 128));
    if (bt) gemm_wmma<true ><<<g, 256, 0, stream>>>(A, Bm, Cm, M, N, K, ldc);
    else    gemm_wmma<false><<<g, 256, 0, stream>>>(A, Bm, Cm, M, N, K, ldc);
  };

  // ---- weight packing (f32 -> K-pair packed f16) ----
  packB(Wq_lat,  wq_lat_p,  DD,  LQ);
  packB(Wkv_lat, wkv_lat_p, DD,  LKV);
  packB(Wrot_q,  wrot_q_p,  LQ,  HH * 2 * ROT);
  packB(Wrot_k,  wrot_k_p,  LKV, HH * 2 * ROT);
  packB(Wq_up,   wq_up_p,   LQ,  DD);
  packB(Wk_up,   wk_up_p,   LKV, DD);
  packB(Wv_up,   wv_up_p,   LKV, DD);
  packB(Wout,    wout_p,    DD,  DD);
  packB(Ws1,     ws1_p,     DD,  2 * FF);
  packB(Ws2,     ws2_p,     FF,  DD);
  for (int e = 0; e < NR; ++e) {
    packB(Wr1 + (size_t)e * DD * 2 * FF, wr1_p + (size_t)e * (DD / 2) * 2 * FF, DD, 2 * FF);
    packB(Wr2 + (size_t)e * FF * DD,     wr2_p + (size_t)e * (FF / 2) * DD,     FF, DD);
  }
  zero_f<<<dim3((TT * DD + 255) / 256), dim3(256), 0, stream>>>(routed_f, TT * DD);

  // ---- sublayer 1: MLA attention ----
  rmsnorm_h<<<dim3(TT), dim3(256), 0, stream>>>(x, norm1_w, xn_h);
  gemm(xn_h, wq_lat_p, zq_f, TT, LQ, DD, LQ, false);
  cvt(zq_f, zq_h, (size_t)TT * LQ);
  gemm(xn_h, wkv_lat_p, zkv_f, TT, LKV, DD, LKV, false);
  cvt(zkv_f, zkv_h, (size_t)TT * LKV);
  gemm(zq_h,  wrot_q_p, qr_f, TT, HH * 2 * ROT, LQ,  HH * 2 * ROT, false);
  gemm(zkv_h, wrot_k_p, kr_f, TT, HH * 2 * ROT, LKV, HH * 2 * ROT, false);
  gemm(zq_h,  wq_up_p,  qc_f, TT, DD, LQ,  DD, false);
  gemm(zkv_h, wk_up_p,  kc_f, TT, DD, LKV, DD, false);
  gemm(zkv_h, wv_up_p,  vc_f, TT, DD, LKV, DD, false);

  {
    const int n = BB * HH * SS * HDIM;
    build_qk<<<dim3((n + 255) / 256), dim3(256), 0, stream>>>(qc_f, qr_f, q_h);
    build_qk<<<dim3((n + 255) / 256), dim3(256), 0, stream>>>(kc_f, kr_f, k_h);
    const int nv = BB * HH * (SS / 2) * HDIM;
    build_v<<<dim3((nv + 255) / 256), dim3(256), 0, stream>>>(vc_f, v_p);
  }

  // per-(b,h): scores = q*k^T -> softmax -> *v, writing head-strided into aout_f
  for (int b = 0; b < BB; ++b) {
    for (int h = 0; h < HH; ++h) {
      const _Float16* qbh = q_h + (size_t)(b * HH + h) * SS * HDIM;
      const unsigned int* kbh = (const unsigned int*)(k_h + (size_t)(b * HH + h) * SS * HDIM);
      const unsigned int* vbh = v_p + (size_t)(b * HH + h) * (SS / 2) * HDIM;
      gemm(qbh, kbh, scores_f, SS, SS, HDIM, SS, true);              // NT: q*k^T
      softmax_causal<<<dim3(SS), dim3(256), 0, stream>>>(scores_f, attn_h);
      float* cdst = aout_f + (size_t)b * SS * DD + h * HDIM;
      gemm(attn_h, vbh, cdst, SS, HDIM, SS, DD, false);              // P*V strided
    }
  }
  cvt(aout_f, ao_h, (size_t)TT * DD);
  gemm(ao_h, wout_p, proj_f, TT, DD, DD, DD, false);
  add2_f<<<dim3((TT * DD + 255) / 256), dim3(256), 0, stream>>>(x, proj_f, xmid_f, TT * DD);

  // ---- sublayer 2: MoE ----
  rmsnorm_h<<<dim3(TT), dim3(256), 0, stream>>>(xmid_f, norm2_w, xn2_h);
  // shared expert
  gemm(xn2_h, ws1_p, hbig_f, TT, 2 * FF, DD, 2 * FF, false);
  swiglu_h<<<dim3((TT * FF + 255) / 256), dim3(256), 0, stream>>>(hbig_f, hsw_h, TT);
  gemm(hsw_h, ws2_p, shared_f, TT, DD, FF, DD, false);
  // routing
  gate_topk<<<dim3(TT), dim3(256), 0, stream>>>(xn2_h, Wgate, ebias, aff_f, top2_i);
  select_capacity<<<dim3(NR), dim3(1024), 0, stream>>>(aff_f, top2_i, wsel_f, sidx_i);
  {
    long long n = (long long)NR * CPAD * DD;
    gather_h<<<dim3((unsigned)((n + 255) / 256)), dim3(256), 0, stream>>>(xn2_h, sidx_i, g_h);
  }
  for (int e = 0; e < NR; ++e) {
    const _Float16* ge  = g_h  + (size_t)e * CPAD * DD;
    const unsigned int* w1e = wr1_p + (size_t)e * (DD / 2) * 2 * FF;
    const unsigned int* w2e = wr2_p + (size_t)e * (FF / 2) * DD;
    gemm(ge, w1e, hbig_f, CPAD, 2 * FF, DD, 2 * FF, false);
    swiglu_h<<<dim3((CPAD * FF + 255) / 256), dim3(256), 0, stream>>>(hbig_f, hsw_h, CPAD);
    gemm(hsw_h, w2e, eout_f, CPAD, DD, FF, DD, false);
    scatter_e<<<dim3((CAP * DD + 255) / 256), dim3(256), 0, stream>>>(eout_f, sidx_i,
                                                                      wsel_f, routed_f, e);
  }

  // ---- final residual combine ----
  add3_f<<<dim3((TT * DD + 255) / 256), dim3(256), 0, stream>>>(xmid_f, shared_f,
                                                                routed_f, out, TT * DD);
}
